// MultiHeadKANAttention_81338090651880
// MI455X (gfx1250) — compile-verified, matrix operand order checked by reading, not run on hardware
//
#include <hip/hip_runtime.h>

// ---------------------------------------------------------------------------
// MultiHeadKANAttention for MI455X (gfx1250, wave32, WMMA).
//
// Roofline: ~130 MB HBM traffic (~5.6 us @ 23.3 TB/s) vs ~116 GFLOP of
// dense-expanded KAN GEMM -> compute-dominated unless routed through the
// matrix pipes. We expand the B-spline basis (only 4 of 35 entries nonzero
// per scalar -> cheap VALU scatter into LDS) into a dense bf16 operand and
// run everything through v_wmma_f32_16x16x32_bf16. The entire bf16 weight
// set (~63 MB) lives in the 192 MB L2, so weight re-reads across M-tiles
// are L2 hits. Weight strips are staged into LDS with CDNA5 async
// global->LDS DMA (ASYNCcnt) so the transfer overlaps the spline-expansion
// VALU work; falls back to synchronous copies if the builtins are absent.
// ---------------------------------------------------------------------------

typedef __attribute__((ext_vector_type(16))) __bf16 v16bf;
typedef __attribute__((ext_vector_type(8)))  float  v8f;
typedef __attribute__((__vector_size__(16))) int    vi4t;   // int4 for async DMA

#define AS1 __attribute__((address_space(1)))
#define AS3 __attribute__((address_space(3)))

#if defined(__AMDGCN__) &&                                        \
    __has_builtin(__builtin_amdgcn_global_load_async_to_lds_b128) && \
    __has_builtin(__builtin_amdgcn_s_wait_asynccnt)
#define USE_ASYNC_LDS 1
#else
#define USE_ASYNC_LDS 0
#endif

#define B_   2
#define L_   1024
#define E_   512
#define H_   8
#define DH   64
#define M_   (B_ * L_)          // 2048 projection rows
#define NB_  35                 // G + K_DEG basis functions
#define WE   40                 // padded expanded width per input element
#define KC   (E_ * WE)          // 20480 expanded K
#define GLO  (-3.0f)
#define GHI  (3.0f)
#define INVH (32.0f / 6.0f)

#define BM   128                // block M tile
#define BN   64                 // block N tile
#define EG   4                  // input elements per K group
#define GK   (EG * WE)          // 160 = 5 * 32

// ---------------------------------------------------------------------------
// Pack [w_base(e,:) ; w_sp(e,0..34,:) ; 0,0,0,0] transposed to bf16:
// wcat[p][n][k], k = e*40 + j. Transposed so WMMA B-fragments are contiguous.
// ---------------------------------------------------------------------------
__global__ __launch_bounds__(256) void prep_weights(
    const float* __restrict__ wqb, const float* __restrict__ wqs,
    const float* __restrict__ wkb, const float* __restrict__ wks,
    const float* __restrict__ wvb, const float* __restrict__ wvs,
    __bf16* __restrict__ wcat)
{
    const int per = E_ * KC;
    int idx = blockIdx.x * blockDim.x + threadIdx.x;
    if (idx >= 3 * per) return;
    int p = idx / per;
    int r = idx - p * per;
    int n = r / KC;
    int k = r - n * KC;
    int e = k / WE;
    int j = k - e * WE;
    const float* wb = (p == 0) ? wqb : (p == 1) ? wkb : wvb;
    const float* ws = (p == 0) ? wqs : (p == 1) ? wks : wvs;
    float v = 0.0f;
    if (j == 0)        v = wb[e * E_ + n];
    else if (j <= NB_) v = ws[(e * NB_ + (j - 1)) * E_ + n];
    wcat[idx] = (__bf16)v;
}

// ---------------------------------------------------------------------------
// KAN projection: Y[M,E] (bf16) = expand(X) @ Wcat, all math on WMMA.
// ---------------------------------------------------------------------------
__global__ __launch_bounds__(256) void kan_project(
    const float*  __restrict__ X,    // [M_, E_] fp32
    const __bf16* __restrict__ Wc,   // [E_, KC] bf16 (n-major, transposed)
    __bf16*       __restrict__ Y)    // [M_, E_] bf16
{
    __shared__ __attribute__((aligned(32))) __bf16 sA[BM][GK];  // 40 KB
    __shared__ __attribute__((aligned(32))) __bf16 sB[BN][GK];  // 20 KB

    const int t     = threadIdx.x;
    const int lane  = t & 31;
    const int wv    = t >> 5;                 // wave 0..7
    const int mBase = blockIdx.x * BM;
    const int nBase = blockIdx.y * BN;

    v8f acc0 = {}, acc1 = {}, acc2 = {}, acc3 = {};

    const int rowA  = (wv << 4) + (lane & 15);
    const int khalf = (lane >> 4) << 4;       // 0 or 16 within the K=32 step

    // B-staging assignment: thread -> (n row, quarter of 160-wide strip)
    const int nStage = t >> 2;                // 0..63
    const int qStage = t & 3;                 // 0..3
    const __bf16* bsrc = Wc + (size_t)(nBase + nStage) * KC + qStage * WE;

    for (int eg = 0; eg < E_ / EG; ++eg) {
        // ---- stage weight strip global -> LDS (16B chunks, all aligned) ----
#if USE_ASYNC_LDS
        {
            // Async DMA: memory -> LDS, overlaps with the spline expansion
            // below; completion enforced via s_wait_asynccnt before barrier.
            const __bf16* g = bsrc + eg * GK;
            __bf16* l = &sB[nStage][qStage * WE];
            #pragma unroll
            for (int i = 0; i < 5; ++i)
                __builtin_amdgcn_global_load_async_to_lds_b128(
                    (AS1 vi4t*)(g + i * 8), (AS3 vi4t*)(l + i * 8), 0, 0);
        }
#else
        {
            const uint4* s = (const uint4*)(bsrc + eg * GK);
            uint4* d = (uint4*)&sB[nStage][qStage * WE];
            #pragma unroll
            for (int i = 0; i < 5; ++i) d[i] = s[i];
            if (eg + 1 < E_ / EG)
                __builtin_prefetch(bsrc + (eg + 1) * GK, 0, 0);
        }
#endif
        // ---- expand activations: silu + 4 nonzero cubic B-spline values ----
        #pragma unroll
        for (int i = 0; i < 2; ++i) {
            int idx = t + (i << 8);           // 0..511
            int row = idx >> 2;               // 0..127
            int el  = idx & 3;                // element within group
            float x = X[(size_t)(mBase + row) * E_ + eg * EG + el];

            uint4 z = {0u, 0u, 0u, 0u};
            uint4* zp = (uint4*)&sA[row][el * WE];
            #pragma unroll
            for (int q = 0; q < 5; ++q) zp[q] = z;

            float s = x / (1.0f + __expf(-x));        // silu, unclamped
            sA[row][el * WE] = (__bf16)s;

            float xc = fminf(fmaxf(x, GLO), GHI - 1e-4f);
            float u  = (xc - GLO) * INVH;
            int   c  = (int)u; c = (c > 31) ? 31 : c; // cell 0..31
            u -= (float)c;
            float um = 1.0f - u;
            float u2 = u * u, u3 = u2 * u;
            float b0 = um * um * um * (1.0f / 6.0f);
            float b1 = (3.0f * u3 - 6.0f * u2 + 4.0f) * (1.0f / 6.0f);
            float b2 = (-3.0f * u3 + 3.0f * u2 + 3.0f * u + 1.0f) * (1.0f / 6.0f);
            float b3 = u3 * (1.0f / 6.0f);
            int o = el * WE + 1 + c;                  // basis c..c+3 active
            sA[row][o + 0] = (__bf16)b0;
            sA[row][o + 1] = (__bf16)b1;
            sA[row][o + 2] = (__bf16)b2;
            sA[row][o + 3] = (__bf16)b3;
        }
#if USE_ASYNC_LDS
        __builtin_amdgcn_s_wait_asynccnt(0);
#endif
        __syncthreads();
        // ---- 5 x (4 WMMA) over the 160-wide expanded chunk ----
        #pragma unroll
        for (int s5 = 0; s5 < 5; ++s5) {
            int kk = s5 * 32 + khalf;
            v16bf a  = *(const v16bf*)&sA[rowA][kk];
            v16bf b0 = *(const v16bf*)&sB[ 0 + (lane & 15)][kk];
            v16bf b1 = *(const v16bf*)&sB[16 + (lane & 15)][kk];
            v16bf b2 = *(const v16bf*)&sB[32 + (lane & 15)][kk];
            v16bf b3 = *(const v16bf*)&sB[48 + (lane & 15)][kk];
            acc0 = __builtin_amdgcn_wmma_f32_16x16x32_bf16(false, a, false, b0, (short)0, acc0, false, false);
            acc1 = __builtin_amdgcn_wmma_f32_16x16x32_bf16(false, a, false, b1, (short)0, acc1, false, false);
            acc2 = __builtin_amdgcn_wmma_f32_16x16x32_bf16(false, a, false, b2, (short)0, acc2, false, false);
            acc3 = __builtin_amdgcn_wmma_f32_16x16x32_bf16(false, a, false, b3, (short)0, acc3, false, false);
        }
        __syncthreads();
    }
    // C layout: VGPR r, lanes 0-15 -> M=r, lanes 16-31 -> M=r+8.
    const int mrow0 = mBase + (wv << 4) + ((lane >> 4) << 3);
    const int col0  = nBase + (lane & 15);
    #pragma unroll
    for (int r = 0; r < 8; ++r) {
        __bf16* yp = Y + (size_t)(mrow0 + r) * E_ + col0;
        yp[ 0] = (__bf16)acc0[r];
        yp[16] = (__bf16)acc1[r];
        yp[32] = (__bf16)acc2[r];
        yp[48] = (__bf16)acc3[r];
    }
}

// ---------------------------------------------------------------------------
// Per (b,h): W2[e][d] = sum_s V[s,e] * K[s,d]  ( = (K^T V)^T, so the next
// kernel's B-operand reads are contiguous).  M=64, N=64, K=1024.
// ---------------------------------------------------------------------------
__global__ __launch_bounds__(128) void vtk_kernel(
    const __bf16* __restrict__ Kb,   // [M_, E_]
    const __bf16* __restrict__ Vb,   // [M_, E_]
    __bf16*       __restrict__ W2)   // [B_*H_, DH, DH]
{
    __shared__ __attribute__((aligned(32))) __bf16 sV[DH][32];  // [e][s]
    __shared__ __attribute__((aligned(32))) __bf16 sK[DH][32];  // [d][s]

    const int t = threadIdx.x, lane = t & 31, wv = t >> 5;
    const int bh = blockIdx.x;            // 0..15
    const int b  = bh >> 3, h = bh & 7;

    v8f acc0 = {}, acc1 = {}, acc2 = {}, acc3 = {};

    const int si = t & 31;                // s within tile
    const int cg = (t >> 5) << 4;         // channel group 0/16/32/48
    const int rowA  = (wv << 4) + (lane & 15);
    const int khalf = (lane >> 4) << 4;

    for (int ks = 0; ks < L_ / 32; ++ks) {
        const int s0 = ks * 32;
        const __bf16* vsrc = Vb + (size_t)(b * L_ + s0 + si) * E_ + h * DH + cg;
        const __bf16* ksrc = Kb + (size_t)(b * L_ + s0 + si) * E_ + h * DH + cg;
        #pragma unroll
        for (int c2 = 0; c2 < 16; ++c2) {   // transpose into LDS
            sV[cg + c2][si] = vsrc[c2];
            sK[cg + c2][si] = ksrc[c2];
        }
        __syncthreads();
        v16bf a  = *(const v16bf*)&sV[rowA][khalf];
        v16bf k0 = *(const v16bf*)&sK[ 0 + (lane & 15)][khalf];
        v16bf k1 = *(const v16bf*)&sK[16 + (lane & 15)][khalf];
        v16bf k2 = *(const v16bf*)&sK[32 + (lane & 15)][khalf];
        v16bf k3 = *(const v16bf*)&sK[48 + (lane & 15)][khalf];
        acc0 = __builtin_amdgcn_wmma_f32_16x16x32_bf16(false, a, false, k0, (short)0, acc0, false, false);
        acc1 = __builtin_amdgcn_wmma_f32_16x16x32_bf16(false, a, false, k1, (short)0, acc1, false, false);
        acc2 = __builtin_amdgcn_wmma_f32_16x16x32_bf16(false, a, false, k2, (short)0, acc2, false, false);
        acc3 = __builtin_amdgcn_wmma_f32_16x16x32_bf16(false, a, false, k3, (short)0, acc3, false, false);
        __syncthreads();
    }
    const int e0 = (wv << 4) + ((lane >> 4) << 3);
    const int d0 = lane & 15;
    #pragma unroll
    for (int r = 0; r < 8; ++r) {
        __bf16* p = W2 + ((size_t)bh * DH + e0 + r) * DH + d0;
        p[ 0] = (__bf16)acc0[r];
        p[16] = (__bf16)acc1[r];
        p[32] = (__bf16)acc2[r];
        p[48] = (__bf16)acc3[r];
    }
}

// ---------------------------------------------------------------------------
// Per (b,h): out[l, h*64+e] = sum_d Q[l,d] * W2[e,d].  A and B fragments load
// straight from global (32B-aligned contiguous v16bf).
// ---------------------------------------------------------------------------
__global__ __launch_bounds__(128) void qw2_kernel(
    const __bf16* __restrict__ Qb,   // [M_, E_]
    const __bf16* __restrict__ W2,   // [B_*H_, DH, DH]
    float*        __restrict__ Out)  // [M_, E_] fp32
{
    const int t = threadIdx.x, lane = t & 31, wv = t >> 5;
    const int bh = blockIdx.x;            // 0..15
    const int b  = bh >> 3, h = bh & 7;
    const int lBase = blockIdx.y * 64 + (wv << 4);

    v8f acc0 = {}, acc1 = {}, acc2 = {}, acc3 = {};

    const int lm    = lBase + (lane & 15);
    const int khalf = (lane >> 4) << 4;
    const __bf16* w2b = W2 + (size_t)bh * DH * DH;

    #pragma unroll
    for (int ks = 0; ks < 2; ++ks) {
        int k0 = ks * 32 + khalf;
        v16bf a  = *(const v16bf*)(Qb + (size_t)(b * L_ + lm) * E_ + h * DH + k0);
        v16bf b0 = *(const v16bf*)(w2b + (size_t)( 0 + (lane & 15)) * DH + k0);
        v16bf b1 = *(const v16bf*)(w2b + (size_t)(16 + (lane & 15)) * DH + k0);
        v16bf b2 = *(const v16bf*)(w2b + (size_t)(32 + (lane & 15)) * DH + k0);
        v16bf b3 = *(const v16bf*)(w2b + (size_t)(48 + (lane & 15)) * DH + k0);
        acc0 = __builtin_amdgcn_wmma_f32_16x16x32_bf16(false, a, false, b0, (short)0, acc0, false, false);
        acc1 = __builtin_amdgcn_wmma_f32_16x16x32_bf16(false, a, false, b1, (short)0, acc1, false, false);
        acc2 = __builtin_amdgcn_wmma_f32_16x16x32_bf16(false, a, false, b2, (short)0, acc2, false, false);
        acc3 = __builtin_amdgcn_wmma_f32_16x16x32_bf16(false, a, false, b3, (short)0, acc3, false, false);
    }
    const int l0 = lBase + ((lane >> 4) << 3);
    const int c0 = h * DH + (lane & 15);
    #pragma unroll
    for (int r = 0; r < 8; ++r) {
        float* p = Out + (size_t)(b * L_ + l0 + r) * E_ + c0;
        p[ 0] = acc0[r];
        p[16] = acc1[r];
        p[32] = acc2[r];
        p[48] = acc3[r];
    }
}

// ---------------------------------------------------------------------------
extern "C" void kernel_launch(void* const* d_in, const int* in_sizes, int n_in,
                              void* d_out, int out_size, void* d_ws, size_t ws_size,
                              hipStream_t stream) {
    const float* q   = (const float*)d_in[0];
    const float* k   = (const float*)d_in[1];
    const float* v   = (const float*)d_in[2];
    const float* wqb = (const float*)d_in[3];
    const float* wqs = (const float*)d_in[4];
    const float* wkb = (const float*)d_in[5];
    const float* wks = (const float*)d_in[6];
    const float* wvb = (const float*)d_in[7];
    const float* wvs = (const float*)d_in[8];
    float* out = (float*)d_out;

    // Workspace layout (bf16): 3 packed weight mats + 3 projections + KtV^T.
    __bf16* ws   = (__bf16*)d_ws;
    const size_t per = (size_t)E_ * KC;               // 10,485,760 bf16 each
    __bf16* Wcat = ws;                                // 3 * per
    __bf16* Qb   = Wcat + 3 * per;                    // M_*E_
    __bf16* Kb   = Qb + (size_t)M_ * E_;
    __bf16* Vb   = Kb + (size_t)M_ * E_;
    __bf16* W2   = Vb + (size_t)M_ * E_;              // 16*64*64

    const int total = 3 * (int)per;
    prep_weights<<<(total + 255) / 256, 256, 0, stream>>>(wqb, wqs, wkb, wks,
                                                          wvb, wvs, Wcat);

    dim3 g1(M_ / BM, E_ / BN);                        // 16 x 8 blocks
    kan_project<<<g1, 256, 0, stream>>>(q, Wcat,           Qb);
    kan_project<<<g1, 256, 0, stream>>>(k, Wcat + per,     Kb);
    kan_project<<<g1, 256, 0, stream>>>(v, Wcat + 2 * per, Vb);

    vtk_kernel<<<B_ * H_, 128, 0, stream>>>(Kb, Vb, W2);
    qw2_kernel<<<dim3(B_ * H_, L_ / 64), 128, 0, stream>>>(Qb, W2, out);
}
